// FP8LinearBase_33157147525258
// MI455X (gfx1250) — matrix operand order repaired and checked by reading, not compiled, not run on hardware
//
#include <hip/hip_runtime.h>
#include <stdint.h>

// Problem constants (fixed by the reference)
#define M_TOT 8192      // B*S = 4*2048
#define K_TOT 4096
#define N_TOT 14336

// Tiling
#define TM 128
#define TN 128
#define TK 32
#define LDH (TK + 8)            // 40 halves = 80B row stride: 16B aligned, bank step 20
#define NSTAGE (K_TOT / TK)     // 128

typedef __bf16 bf16_t;
typedef bf16_t v16bf __attribute__((ext_vector_type(16)));
typedef float  v8f   __attribute__((ext_vector_type(8)));
typedef unsigned int v2u __attribute__((ext_vector_type(2)));
typedef unsigned int v4u __attribute__((ext_vector_type(4)));
typedef unsigned int v8u __attribute__((ext_vector_type(8)));
typedef int v4i __attribute__((ext_vector_type(4)));
typedef int v8i __attribute__((ext_vector_type(8)));
typedef unsigned short u16;
typedef unsigned int u32;
typedef unsigned long long u64;

#if __has_builtin(__builtin_amdgcn_tensor_load_to_lds) && __has_builtin(__builtin_amdgcn_s_wait_tensorcnt)
#define USE_TDM 1
#else
#define USE_TDM 0
#endif

static __device__ __forceinline__ u16 f32_to_bf16_rne(float f) {
    unsigned int u = __float_as_uint(f);
    unsigned int r = (u + 0x7fffu + ((u >> 16) & 1u)) >> 16;
    return (u16)r;
}

// Dequant two e4m3 bytes (byte e of qa -> k, byte e of qb -> k+1) into one
// packed bf16 dword. v_perm places the bytes at bits [15:8] / [31:24]; the
// sign+exp/mant slide forms bf16 with fp8 exponent bias; one v_pk_mul_bf16
// by 2^120 (0x7B80) rebiases both halves. Exact incl. fp8 denormals.
static __device__ __forceinline__ u32 fp8pair_to_bf16pair(u32 qa, u32 qb, int e) {
    const u32 sel = ((u32)e << 24) | 0x000C0000u | ((u32)(4 + e) << 8) | 0x0000000Cu;
    const u32 t   = __builtin_amdgcn_perm(qa, qb, sel);          // [qb_e, 0, qa_e, 0]
    const u32 pre = (t & 0x80008000u) | ((t >> 4) & 0x07F007F0u);
    u32 d;
    asm("v_pk_mul_bf16 %0, %1, %2" : "=v"(d) : "v"(pre), "v"(0x7B807B80u));
    return d;
}

static __device__ __forceinline__ v16bf ld_frag(const u16* p0, const u16* p1) {
    v4u lo = *(const v4u*)p0;
    v4u hi = *(const v4u*)p1;
    v8u c;
    c[0] = lo[0]; c[1] = lo[1]; c[2] = lo[2]; c[3] = lo[3];
    c[4] = hi[0]; c[5] = hi[1]; c[6] = hi[2]; c[7] = hi[3];
    return __builtin_bit_cast(v16bf, c);
}

#if USE_TDM
// TDM descriptor: 2D tile TK x TM halves out of the [M_TOT, K_TOT] bf16 tensor,
// hardware-padded 4 dwords after every 16 dwords (64B row -> 80B LDS stride).
static __device__ __forceinline__ void tdm_load_A_tile(const u16* gsrc, u32 lds_byte) {
    const u64 ga = (u64)(uintptr_t)gsrc;
    v4u g0;
    g0[0] = 1u;                                            // count=1, user mode
    g0[1] = lds_byte;                                      // lds_addr (WG-relative bytes)
    g0[2] = (u32)ga;                                       // global_addr[31:0]
    g0[3] = ((u32)(ga >> 32) & 0x01ffffffu) | (2u << 30);  // global_addr[56:32] | type=2
    v8i g1;
    g1[0] = (int)((1u << 16)      // data_size = 2B
                | (1u << 20)      // pad_enable
                | (3u << 22)      // pad_interval: every 16 dwords (one 64B row)
                | (3u << 25));    // pad_amount: 4 dwords (16B)
    g1[1] = (int)(((u32)K_TOT & 0xffffu) << 16);                        // tensor_dim0[15:0]
    g1[2] = (int)(((u32)K_TOT >> 16) | (((u32)M_TOT & 0xffffu) << 16)); // dim0 hi | dim1 lo
    g1[3] = (int)(((u32)M_TOT >> 16) | ((u32)TK << 16));                // dim1 hi | tile_dim0=32
    g1[4] = (int)TM;                                                    // tile_dim1=128, tile_dim2=0
    g1[5] = (int)K_TOT;                                                 // tensor_dim0_stride = 4096
    g1[6] = 0;                                                          // stride hi | dim1_stride
    g1[7] = 0;
    v4i g2; g2[0] = 1; g2[1] = 1; g2[2] = 0; g2[3] = 0;    // tensor_dim2=1, tensor_dim3=1
    v4i g3; g3[0] = 0; g3[1] = (int)(1u << 16); g3[2] = 0; g3[3] = 0;  // tensor_dim4=1
    v8i gx; gx[0]=0; gx[1]=0; gx[2]=0; gx[3]=0; gx[4]=0; gx[5]=0; gx[6]=0; gx[7]=0;
    // 6-arg form (clang-23 / therock-10.0 headers): groups 0-3, extra group, cpol
    __builtin_amdgcn_tensor_load_to_lds(g0, g1, g2, g3, gx, 0);
}
#endif

__global__ __launch_bounds__(256)
void fp8_linear_wmma_kernel(const u16* __restrict__ X,        // bf16 bits [M, K]
                            const uint8_t* __restrict__ Wq,   // fp8 e4m3  [K, N]
                            const float* __restrict__ scale,  // [N]
                            const float* __restrict__ bias,   // [N]
                            u16* __restrict__ Out)            // bf16 bits [M, N]
{
    __shared__ u16 shA[2][TM * LDH];   // [m][k] bf16, double buffered (10KB each)
    __shared__ u16 shB[2][TN * LDH];   // [n][k] bf16 (transposed weight tile)

    const int tid  = threadIdx.x;
    const int lane = tid & 31;
    const int wave = tid >> 5;
    const int wm   = wave >> 2;          // 0..1 : 64 M-rows per wave row
    const int wn   = wave & 3;           // 0..3 : 32 N-cols per wave col
    const int l16  = lane & 15;
    const int half = (lane < 16) ? 0 : 1;

    const int tileM = blockIdx.y * TM;
    const int tileN = blockIdx.x * TN;

    // B staging map: thread -> k-pair (2 rows) x 8 n's : 16 bytes per stage
    const int b_kp = tid >> 4;           // 0..15 -> k = 2*b_kp, 2*b_kp+1
    const int b_n  = (tid & 15) * 8;     // 0..120

#if !USE_TDM
    // A staging map (fallback): 2 chunks of 16B per thread
    const int a_row = tid >> 2;          // 0..63
    const int a_col = (tid & 3) * 8;     // halves
#endif

    v8f acc[4][2];
    const v8f vzero = {0.f, 0.f, 0.f, 0.f, 0.f, 0.f, 0.f, 0.f};
#pragma unroll
    for (int i = 0; i < 4; ++i)
#pragma unroll
        for (int j = 0; j < 2; ++j)
            acc[i][j] = vzero;

    // ---------------- prologue: fill stage 0 ----------------
    {
        const size_t g = (size_t)(2 * b_kp) * N_TOT + tileN + b_n;
        const v2u qa = *(const v2u*)(Wq + g);
        const v2u qb = *(const v2u*)(Wq + g + N_TOT);
#pragma unroll
        for (int e = 0; e < 8; ++e) {
            *(u32*)&shB[0][(b_n + e) * LDH + 2 * b_kp] =
                fp8pair_to_bf16pair(qa[e >> 2], qb[e >> 2], e & 3);
        }
#if USE_TDM
        if (wave == 0) {
            tdm_load_A_tile(X + (size_t)tileM * K_TOT, (u32)(uintptr_t)&shA[0][0]);
            __builtin_amdgcn_s_wait_tensorcnt(0);
        }
#else
        const v4u a0 = *(const v4u*)(X + (size_t)(tileM + a_row) * K_TOT + a_col);
        const v4u a1 = *(const v4u*)(X + (size_t)(tileM + 64 + a_row) * K_TOT + a_col);
        *(v4u*)&shA[0][a_row * LDH + a_col] = a0;
        *(v4u*)&shA[0][(64 + a_row) * LDH + a_col] = a1;
#endif
    }
    __syncthreads();

    // ---------------- main pipelined loop ----------------
    for (int s = 0; s < NSTAGE; ++s) {
        const int cur = s & 1;
        const int nxt = cur ^ 1;
        const bool pref = (s + 1) < NSTAGE;
        const int k0n = (s + 1) * TK;

        // --- issue next-stage loads (global B to regs, A via TDM / regs) ---
        v2u qa = {0u, 0u}, qb = {0u, 0u};
#if !USE_TDM
        v4u a0 = {0u,0u,0u,0u}, a1 = {0u,0u,0u,0u};
#endif
        if (pref) {
            const size_t g = (size_t)(k0n + 2 * b_kp) * N_TOT + tileN + b_n;
            qa = *(const v2u*)(Wq + g);
            qb = *(const v2u*)(Wq + g + N_TOT);
#if USE_TDM
            if (wave == 0)
                tdm_load_A_tile(X + (size_t)tileM * K_TOT + k0n,
                                (u32)(uintptr_t)&shA[nxt][0]);
#else
            a0 = *(const v4u*)(X + (size_t)(tileM + a_row) * K_TOT + k0n + a_col);
            a1 = *(const v4u*)(X + (size_t)(tileM + 64 + a_row) * K_TOT + k0n + a_col);
#endif
        }

        // --- compute current stage: 8 WMMAs ---
        {
            v16bf afr[4];
#pragma unroll
            for (int i = 0; i < 4; ++i) {
                const u16* ab = &shA[cur][(wm * 64 + i * 16 + l16) * LDH + half * 8];
                afr[i] = ld_frag(ab, ab + 16);
            }
            v16bf bfr[2];
#pragma unroll
            for (int j = 0; j < 2; ++j) {
                const u16* bb = &shB[cur][(wn * 32 + j * 16 + l16) * LDH + half * 8];
                bfr[j] = ld_frag(bb, bb + 16);
            }
#pragma unroll
            for (int i = 0; i < 4; ++i)
#pragma unroll
                for (int j = 0; j < 2; ++j)
                    acc[i][j] = __builtin_amdgcn_wmma_f32_16x16x32_bf16(
                        false, afr[i], false, bfr[j], (short)0, acc[i][j], false, false);
        }

        // --- dequant + publish next stage to LDS ---
        if (pref) {
#pragma unroll
            for (int e = 0; e < 8; ++e) {
                *(u32*)&shB[nxt][(b_n + e) * LDH + 2 * b_kp] =
                    fp8pair_to_bf16pair(qa[e >> 2], qb[e >> 2], e & 3);
            }
#if !USE_TDM
            *(v4u*)&shA[nxt][a_row * LDH + a_col] = a0;
            *(v4u*)&shA[nxt][(64 + a_row) * LDH + a_col] = a1;
#endif
        }
#if USE_TDM
        if (pref && wave == 0) __builtin_amdgcn_s_wait_tensorcnt(0);
#endif
        __syncthreads();
    }

    // ---------------- epilogue: fused scale + bias, bf16 store ----------------
#pragma unroll
    for (int j = 0; j < 2; ++j) {
        const int col = tileN + wn * 32 + j * 16 + l16;
        const float sc = scale[col];
        const float bs = bias[col];
#pragma unroll
        for (int i = 0; i < 4; ++i) {
            const int rowbase = tileM + wm * 64 + i * 16 + half * 8;
#pragma unroll
            for (int r = 0; r < 8; ++r) {
                const float v = acc[i][j][r] * sc + bs;
                Out[(size_t)(rowbase + r) * N_TOT + col] = f32_to_bf16_rne(v);
            }
        }
    }
}

extern "C" void kernel_launch(void* const* d_in, const int* in_sizes, int n_in,
                              void* d_out, int out_size, void* d_ws, size_t ws_size,
                              hipStream_t stream) {
    const u16*     X  = (const u16*)d_in[0];        // bf16 [4,2048,4096]
    const uint8_t* Wq = (const uint8_t*)d_in[1];    // fp8  [4096,14336]
    const float*   sc = (const float*)d_in[2];      // [14336]
    const float*   bs = (const float*)d_in[3];      // [14336]
    u16*           O  = (u16*)d_out;                // bf16 [4,2048,14336]

    dim3 grid(N_TOT / TN, M_TOT / TM);   // 112 x 64 workgroups
    dim3 block(256);                     // 8 wave32
    fp8_linear_wmma_kernel<<<grid, block, 0, stream>>>(X, Wq, sc, bs, O);
}
